// NCMultiAgentPolicy_50328426774559
// MI455X (gfx1250) — matrix operand order repaired and checked by reading, not compiled, not run on hardware
//
#include <hip/hip_runtime.h>
#include <hip/hip_bf16.h>

// ---------------------------------------------------------------------------
// NCMultiAgentPolicy for MI455X (gfx1250, wave32, WMMA).
//
// Sizes: T=2048, N=128, N_S=64, N_A=8, N_FC=128, N_H=128, N_N=2 (ring).
// Strategy:
//   1) Hoist h-independent encoders x,p out of the scan -> per-agent bf16 WMMA
//      GEMMs over all timesteps (xp_all).
//   2) Persistent 8-workgroup scan kernel (16 agents/WG). Per step:
//      m matvec (VALU, bf16 weights), Wh GEMM via v_wmma_f32_16x16x32_bf16,
//      ring-GAT softmax, LSTM gate matvecs (bf16 weights, f32 accum).
//      2 grid syncs/step via atomics. Pure-copy LDS fills use
//      GLOBAL_LOAD_ASYNC_TO_LDS_B128 (ASYNCcnt) when the toolchain has it.
//   3) Heads: per-agent WMMA GEMM for logits (+N padded 8->16), scalar dot
//      for the critic value.
// Workspace: ~400 MB.
// ---------------------------------------------------------------------------

#define TT 2048
#define NN 128
#define NSZ 64
#define NAA 8
#define NFC 128
#define NH 128

#define SCAN_WG 8
#define AGW 16      // agents per scan workgroup

typedef __attribute__((ext_vector_type(16))) __bf16 v16bf;
typedef __attribute__((ext_vector_type(8)))  __bf16 v8bf;
typedef __attribute__((ext_vector_type(8)))  float  v8f;
typedef __attribute__((ext_vector_type(4)))  int    v4i;

// ---------------- async global->LDS (CDNA5) ---------------------------------
#if defined(__has_builtin)
#if __has_builtin(__builtin_amdgcn_global_load_async_to_lds_b128) && \
    __has_builtin(__builtin_amdgcn_s_wait_asynccnt)
#define USE_ASYNC_LDS 1
#endif
#endif
#ifndef USE_ASYNC_LDS
#define USE_ASYNC_LDS 0
#endif

#if USE_ASYNC_LDS
__device__ __forceinline__ void async_cp_b128(void* lds, const void* gsrc) {
  // 16B global -> LDS, tracked by ASYNCcnt, no VGPR round trip.
  // Builtin params are v4i* in the device/shared language address spaces;
  // clang inserts the generic->AS addrspacecast for builtin args, so pass
  // plain generic v4i* (C-style cast also drops the const).
  __builtin_amdgcn_global_load_async_to_lds_b128((v4i*)gsrc, (v4i*)lds, 0, 0);
}
__device__ __forceinline__ void async_wait() {
  __builtin_amdgcn_s_wait_asynccnt(0);    // s_wait_asynccnt 0
}
#endif

// ---------------- WMMA helpers (16x16x32 bf16, wave32) ----------------------
// A (16x32, row-major source, lda in halfs): lane L holds row (L&15);
//   lanes 0-15: K = k0+{0..15}; lanes 16-31: K = k0+{16..31}.
__device__ __forceinline__ v16bf load_a16x32(const __bf16* A, int lda, int row0, int k0) {
  int lane = threadIdx.x & 31;
  int r = row0 + (lane & 15);
  int kb = k0 + ((lane >> 4) << 4);
  const __bf16* p = A + (long)r * lda + kb;
  v8bf lo = *(const v8bf*)p;
  v8bf hi = *(const v8bf*)(p + 8);
  v16bf a;
#pragma unroll
  for (int j = 0; j < 8; ++j) { a[j] = lo[j]; a[8 + j] = hi[j]; }
  return a;
}

// B tiles are pre-swizzled: tile = 512 halfs, per-lane contiguous 16 halfs:
//   element (lane, i):  k = kt*32 + ((lane>=16)?16:0) + i,  col = nt*16 + (lane&15)
__device__ __forceinline__ v16bf load_b_swz(const __bf16* Bsw, int tile) {
  int lane = threadIdx.x & 31;
  return *(const v16bf*)(Bsw + (long)tile * 512 + lane * 16);
}

__device__ __forceinline__ v8f wmma_bf16(v16bf a, v16bf b, v8f c) {
  return __builtin_amdgcn_wmma_f32_16x16x32_bf16(false, a, false, b, (short)0, c, false, false);
}

// C/D element r: row = r + ((lane>=16)?8:0), col = lane&15.

// ---------------- staging kernels -------------------------------------------
__global__ void k_stage_xin(const float* __restrict__ obs, __bf16* __restrict__ xin) {
  long total = (long)NN * TT * 192;
  for (long i = blockIdx.x * (long)blockDim.x + threadIdx.x; i < total;
       i += (long)gridDim.x * blockDim.x) {
    int k = (int)(i % 192);
    long r = i / 192;
    int t = (int)(r % TT);
    int n = (int)(r / TT);
    int sn, kk;
    if (k < 64)       { sn = n;                     kk = k; }
    else if (k < 128) { sn = (n + NN - 1) & (NN-1); kk = k - 64; }
    else              { sn = (n + 1) & (NN-1);      kk = k - 128; }
    xin[i] = (__bf16)obs[((long)t * NN + sn) * NSZ + kk];
  }
}

__global__ void k_stage_pin(const float* __restrict__ fps, __bf16* __restrict__ pin) {
  long total = (long)NN * TT * 32;
  for (long i = blockIdx.x * (long)blockDim.x + threadIdx.x; i < total;
       i += (long)gridDim.x * blockDim.x) {
    int k = (int)(i % 32);
    long r = i / 32;
    int t = (int)(r % TT);
    int n = (int)(r / TT);
    float v = 0.f;
    if (k < 8)        v = fps[((long)t * NN + ((n + NN - 1) & (NN-1))) * NAA + k];
    else if (k < 16)  v = fps[((long)t * NN + ((n + 1) & (NN-1))) * NAA + (k - 8)];
    pin[i] = (__bf16)v;
  }
}

// B-swizzles: layout [..tiles..][lane(32)][i(16)]
__global__ void k_swz_fcx(const float* __restrict__ W, __bf16* __restrict__ B) {
  long total = (long)NN * 6 * 8 * 512;
  for (long idx = blockIdx.x * (long)blockDim.x + threadIdx.x; idx < total;
       idx += (long)gridDim.x * blockDim.x) {
    int i = (int)(idx & 15);
    int lane = (int)((idx >> 4) & 31);
    int nt = (int)((idx >> 9) & 7);
    int kt = (int)((idx >> 12) % 6);
    int n  = (int)(idx / 24576);
    int k = kt * 32 + ((lane >> 4) << 4) + i;      // < 192
    int col = nt * 16 + (lane & 15);
    B[idx] = (__bf16)W[((long)n * 192 + k) * 128 + col];
  }
}

__global__ void k_swz_fcp(const float* __restrict__ W, __bf16* __restrict__ B) {
  long total = (long)NN * 8 * 512;
  for (long idx = blockIdx.x * (long)blockDim.x + threadIdx.x; idx < total;
       idx += (long)gridDim.x * blockDim.x) {
    int i = (int)(idx & 15);
    int lane = (int)((idx >> 4) & 31);
    int nt = (int)((idx >> 9) & 7);
    int n  = (int)(idx / 4096);
    int k = ((lane >> 4) << 4) + i;                // pad K 16 -> 32 with zeros
    int col = nt * 16 + (lane & 15);
    float v = (k < 16) ? W[((long)n * 16 + k) * 128 + col] : 0.f;
    B[idx] = (__bf16)v;
  }
}

__global__ void k_swz_gat(const float* __restrict__ W, __bf16* __restrict__ B) {
  long total = (long)12 * 24 * 512;
  for (long idx = blockIdx.x * (long)blockDim.x + threadIdx.x; idx < total;
       idx += (long)gridDim.x * blockDim.x) {
    int i = (int)(idx & 15);
    int lane = (int)((idx >> 4) & 31);
    int nt = (int)((idx >> 9) % 24);
    int kt = (int)(idx / (512 * 24));
    int k = kt * 32 + ((lane >> 4) << 4) + i;
    int col = nt * 16 + (lane & 15);
    B[idx] = (__bf16)W[(long)k * 384 + col];
  }
}

__global__ void k_swz_act(const float* __restrict__ W, __bf16* __restrict__ B) {
  long total = (long)NN * 4 * 512;
  for (long idx = blockIdx.x * (long)blockDim.x + threadIdx.x; idx < total;
       idx += (long)gridDim.x * blockDim.x) {
    int i = (int)(idx & 15);
    int lane = (int)((idx >> 4) & 31);
    int kt = (int)((idx >> 9) & 3);
    int n  = (int)(idx / 2048);
    int k = kt * 32 + ((lane >> 4) << 4) + i;
    int col = lane & 15;                            // pad N 8 -> 16 with zeros
    float v = (col < 8) ? W[((long)n * 128 + k) * 8 + col] : 0.f;
    B[idx] = (__bf16)v;
  }
}

__global__ void k_cvt_bf16(const float* __restrict__ s, __bf16* __restrict__ d, long n) {
  for (long i = blockIdx.x * (long)blockDim.x + threadIdx.x; i < n;
       i += (long)gridDim.x * blockDim.x)
    d[i] = (__bf16)s[i];
}

// ---------------- batched x/p encoder GEMMs (parallel over T) ---------------
// Per agent n: x = relu([2048x192]@[192x128]+b), p = relu([2048x32p]@[32x128]+b)
// xp layout: [N][T][256] bf16  (x | p)
__global__ void k_gemm_xp(const __bf16* __restrict__ xin, const __bf16* __restrict__ pin,
                          const __bf16* __restrict__ WfcxB, const __bf16* __restrict__ WfcpB,
                          const float* __restrict__ b_fcx, const float* __restrict__ b_fcp,
                          __bf16* __restrict__ xp) {
  int n = blockIdx.x;
  int t0 = blockIdx.y * 128;
  int wave = threadIdx.x >> 5, lane = threadIdx.x & 31;
  int row0 = t0 + wave * 16;
  int col = lane & 15, mb = (lane >> 4) << 3;

  const __bf16* A = xin + (long)n * TT * 192;
  v16bf a[6];
#pragma unroll
  for (int kt = 0; kt < 6; ++kt) a[kt] = load_a16x32(A, 192, row0, kt * 32);

  const __bf16* Bx = WfcxB + (long)n * 6 * 8 * 512;
  for (int nt = 0; nt < 8; ++nt) {
    v8f c = {};
#pragma unroll
    for (int kt = 0; kt < 6; ++kt)
      c = wmma_bf16(a[kt], load_b_swz(Bx, kt * 8 + nt), c);
    float bias = b_fcx[n * 128 + nt * 16 + col];
#pragma unroll
    for (int r = 0; r < 8; ++r) {
      float v = c[r] + bias; v = v > 0.f ? v : 0.f;
      int t = row0 + mb + r;
      xp[((long)n * TT + t) * 256 + nt * 16 + col] = (__bf16)v;
    }
  }

  v16bf ap = load_a16x32(pin + (long)n * TT * 32, 32, row0, 0);
  const __bf16* Bp = WfcpB + (long)n * 8 * 512;
  for (int nt = 0; nt < 8; ++nt) {
    v8f c = {};
    c = wmma_bf16(ap, load_b_swz(Bp, nt), c);
    float bias = b_fcp[n * 128 + nt * 16 + col];
#pragma unroll
    for (int r = 0; r < 8; ++r) {
      float v = c[r] + bias; v = v > 0.f ? v : 0.f;
      int t = row0 + mb + r;
      xp[((long)n * TT + t) * 256 + 128 + nt * 16 + col] = (__bf16)v;
    }
  }
}

// ---------------- grid sync (persistent scan) -------------------------------
__device__ __forceinline__ void gsync(unsigned* cnt, unsigned* bar) {
  __syncthreads();
  __threadfence();
  if (threadIdx.x == 0) {
    unsigned target = (++(*bar)) * SCAN_WG;
    __hip_atomic_fetch_add(cnt, 1u, __ATOMIC_RELEASE, __HIP_MEMORY_SCOPE_AGENT);
    while (__hip_atomic_load(cnt, __ATOMIC_ACQUIRE, __HIP_MEMORY_SCOPE_AGENT) < target)
      __builtin_amdgcn_s_sleep(2);
  }
  __syncthreads();
  __threadfence();
}

__global__ void k_init_sync(unsigned* sync) { if (threadIdx.x == 0) *sync = 0u; }

// ---------------- the recurrent scan ----------------------------------------
__global__ void __launch_bounds__(256)
k_scan(const __bf16* __restrict__ xp, const __bf16* __restrict__ Wfcm_bf,
       const float* __restrict__ b_fcm, const __bf16* __restrict__ WgatB,
       const float* __restrict__ a_src, const float* __restrict__ a_dst,
       const __bf16* __restrict__ Wih_bf, const __bf16* __restrict__ Whh_bf,
       const float* __restrict__ b_lstm, const float* __restrict__ dones,
       const float* __restrict__ states0, float* __restrict__ curh,
       float* __restrict__ Whg, float* __restrict__ dstg,
       unsigned* __restrict__ sync, __bf16* __restrict__ hsb) {
  __shared__ float  h_lds[AGW + 2][NH];     // rows: A0-1 .. A0+16 (masked h)
  __shared__ float  c_lds[AGW][NH];
  __shared__ __bf16 sfeat[AGW][384];        // x | p | m  (GEMM A operand)
  __shared__ float  wh[AGW][384];
  __shared__ float  whext[2][384];
  __shared__ float  sgat[AGW][384];
  __shared__ float  gates[AGW][512];
  __shared__ float  s_src[AGW];
  __shared__ float  s_dst[AGW + 2];
  __shared__ float  attw[AGW][3];
  __shared__ unsigned bar_s;

  const int wg = blockIdx.x;
  const int A0 = wg * AGW;
  const int tid = threadIdx.x;
  if (tid == 0) bar_s = 0u;
  __syncthreads();

  // init h/c from states0
  for (int o = tid; o < AGW * NH; o += 256) {
    int a = o >> 7, d = o & 127;
    c_lds[a][d] = states0[(A0 + a) * 256 + 128 + d];
    curh[(A0 + a) * NH + d] = states0[(A0 + a) * 256 + d];
  }
  gsync(sync, &bar_s);

  for (int t = 0; t < TT; ++t) {
    // x | p from precomputed xp_all: pure copy -> async DMA to LDS
#if USE_ASYNC_LDS
    for (int o = tid; o < AGW * 32; o += 256) {       // 16 rows x 32 chunks x 16B
      int a = o >> 5, cc = o & 31;
      async_cp_b128(&sfeat[a][cc * 8],
                    xp + ((long)(A0 + a) * TT + t) * 256 + cc * 8);
    }
#else
    for (int o = tid; o < AGW * 256; o += 256) {
      int a = o >> 8, f = o & 255;
      sfeat[a][f] = xp[((long)(A0 + a) * TT + t) * 256 + f];
    }
#endif
    // masked h (own + ring-neighbor rows), mask c  (transform -> scalar path)
    for (int o = tid; o < (AGW + 2) * NH; o += 256) {
      int r = o >> 7, d = o & 127;
      int ga = (A0 - 1 + r + NN) & (NN - 1);
      h_lds[r][d] = curh[ga * NH + d] * (1.0f - dones[(long)t * NN + ga]);
    }
    for (int o = tid; o < AGW * NH; o += 256) {
      int a = o >> 7, d = o & 127;
      c_lds[a][d] *= (1.0f - dones[(long)t * NN + A0 + a]);
    }
#if USE_ASYNC_LDS
    async_wait();
#endif
    __syncthreads();
    // m = relu([h_left|h_right] . W_fcm[n] + b)
    for (int o = tid; o < AGW * 128; o += 256) {
      int a = o >> 7, g = o & 127;
      const __bf16* w = Wfcm_bf + (long)(A0 + a) * 256 * 128 + g;
      float acc = b_fcm[(A0 + a) * 128 + g];
#pragma unroll 4
      for (int k = 0; k < 128; ++k) acc += h_lds[a][k] * (float)w[(long)k * 128];
#pragma unroll 4
      for (int k = 0; k < 128; ++k) acc += h_lds[a + 2][k] * (float)w[(long)(128 + k) * 128];
      sfeat[a][256 + g] = (__bf16)(acc > 0.f ? acc : 0.f);
    }
    __syncthreads();
    // Wh = sfeat(16x384) @ W_gat(384x384) via WMMA; wave w -> ntiles 3w..3w+2
    {
      int wave = tid >> 5, lane = tid & 31;
      int col = lane & 15, mb = (lane >> 4) << 3;
      int nt0 = wave * 3;
      v8f c0 = {}, c1 = {}, c2 = {};
      for (int kt = 0; kt < 12; ++kt) {
        v16bf a = load_a16x32(&sfeat[0][0], 384, 0, kt * 32);
        c0 = wmma_bf16(a, load_b_swz(WgatB, kt * 24 + nt0 + 0), c0);
        c1 = wmma_bf16(a, load_b_swz(WgatB, kt * 24 + nt0 + 1), c1);
        c2 = wmma_bf16(a, load_b_swz(WgatB, kt * 24 + nt0 + 2), c2);
      }
#pragma unroll
      for (int r = 0; r < 8; ++r) {
        int m = mb + r;
        wh[m][(nt0 + 0) * 16 + col] = c0[r];
        wh[m][(nt0 + 1) * 16 + col] = c1[r];
        wh[m][(nt0 + 2) * 16 + col] = c2[r];
        Whg[(A0 + m) * 384 + (nt0 + 0) * 16 + col] = c0[r];
        Whg[(A0 + m) * 384 + (nt0 + 1) * 16 + col] = c1[r];
        Whg[(A0 + m) * 384 + (nt0 + 2) * 16 + col] = c2[r];
      }
    }
    __syncthreads();
    // attention scalars
    if (tid < AGW) {
      float ds = 0.f, dd = 0.f;
      for (int f = 0; f < 384; ++f) { float v = wh[tid][f]; ds += v * a_src[f]; dd += v * a_dst[f]; }
      s_src[tid] = ds; s_dst[tid + 1] = dd;
      dstg[A0 + tid] = dd;
    }
    gsync(sync, &bar_s);                    // Wh/dst globally visible
    // neighbor Wh rows: pure copy -> async DMA to LDS
#if USE_ASYNC_LDS
    for (int o = tid; o < 2 * 96; o += 256) {         // 2 rows x 96 chunks x 16B
      int e = o / 96, cc = o % 96;
      int ga = e == 0 ? ((A0 - 1 + NN) & (NN - 1)) : ((A0 + AGW) & (NN - 1));
      async_cp_b128(&whext[e][cc * 4], Whg + ga * 384 + cc * 4);
    }
#else
    for (int o = tid; o < 2 * 384; o += 256) {
      int e = o / 384, f = o % 384;
      int ga = e == 0 ? ((A0 - 1 + NN) & (NN - 1)) : ((A0 + AGW) & (NN - 1));
      whext[e][f] = Whg[ga * 384 + f];
    }
#endif
    if (tid == 0) s_dst[0] = dstg[(A0 - 1 + NN) & (NN - 1)];
    if (tid == 1) s_dst[AGW + 1] = dstg[(A0 + AGW) & (NN - 1)];
#if USE_ASYNC_LDS
    async_wait();
#endif
    __syncthreads();
    // ring-GAT softmax over {i-1, i, i+1}
    if (tid < AGW) {
      float si = s_src[tid];
      float el = si + s_dst[tid];
      float es = si + s_dst[tid + 1];
      float er = si + s_dst[tid + 2];
      el = el > 0.f ? el : 0.2f * el;
      es = es > 0.f ? es : 0.2f * es;
      er = er > 0.f ? er : 0.2f * er;
      float mx = fmaxf(el, fmaxf(es, er));
      float wl = __expf(el - mx), ws = __expf(es - mx), wr = __expf(er - mx);
      float inv = 1.0f / (wl + ws + wr);
      attw[tid][0] = wl * inv; attw[tid][1] = ws * inv; attw[tid][2] = wr * inv;
    }
    __syncthreads();
    // s_gat = elu(attn @ Wh)
    for (int o = tid; o < AGW * 384; o += 256) {
      int a = o / 384, f = o % 384;
      const float* wl = (a == 0) ? whext[0] : wh[a - 1];
      const float* wr = (a == AGW - 1) ? whext[1] : wh[a + 1];
      float v = attw[a][0] * wl[f] + attw[a][1] * wh[a][f] + attw[a][2] * wr[f];
      sgat[a][f] = v > 0.f ? v : (__expf(v) - 1.0f);
    }
    __syncthreads();
    // gates = s_gat.W_ih[n] + h.W_hh[n] + b
    for (int o = tid; o < AGW * 512; o += 256) {
      int a = o >> 9, g = o & 511;
      int ga = A0 + a;
      float acc = b_lstm[ga * 512 + g];
      const __bf16* wi = Wih_bf + (long)ga * 384 * 512 + g;
      __builtin_prefetch((const void*)(wi + (long)64 * 512), 0, 1);
#pragma unroll 4
      for (int f = 0; f < 384; ++f) acc += sgat[a][f] * (float)wi[(long)f * 512];
      const __bf16* whw = Whh_bf + (long)ga * 128 * 512 + g;
#pragma unroll 4
      for (int k = 0; k < 128; ++k) acc += h_lds[a + 1][k] * (float)whw[(long)k * 512];
      gates[a][g] = acc;
    }
    __syncthreads();
    // LSTM pointwise
    for (int o = tid; o < AGW * NH; o += 256) {
      int a = o >> 7, d = o & 127;
      float gi = gates[a][d], gf = gates[a][128 + d];
      float gg = gates[a][256 + d], go = gates[a][384 + d];
      float si = 1.0f / (1.0f + __expf(-gi));
      float sf = 1.0f / (1.0f + __expf(-gf));
      float so = 1.0f / (1.0f + __expf(-go));
      float cn = sf * c_lds[a][d] + si * tanhf(gg);
      float hn = so * tanhf(cn);
      c_lds[a][d] = cn;
      curh[(A0 + a) * NH + d] = hn;
      hsb[((long)(A0 + a) * TT + t) * NH + d] = (__bf16)hn;
    }
    gsync(sync, &bar_s);                    // h globally visible
  }
}

// ---------------- heads (parallel over T) -----------------------------------
__global__ void k_heads(const __bf16* __restrict__ hsb, const __bf16* __restrict__ WactB,
                        const float* __restrict__ b_act, const float* __restrict__ W_cri,
                        const float* __restrict__ b_cri, const int* __restrict__ acts,
                        float* __restrict__ out) {
  int n = blockIdx.x;
  int t0 = blockIdx.y * 256;
  int wave = threadIdx.x >> 5, lane = threadIdx.x & 31;
  int col = lane & 15, mb = (lane >> 4) << 3;
  const __bf16* A = hsb + (long)n * TT * NH;
  const __bf16* B = WactB + (long)n * 4 * 512;

  for (int mt = 0; mt < 2; ++mt) {
    int row0 = t0 + (wave * 2 + mt) * 16;
    v8f c = {};
#pragma unroll
    for (int kt = 0; kt < 4; ++kt)
      c = wmma_bf16(load_a16x32(A, NH, row0, kt * 32), load_b_swz(B, kt), c);
    if (col < 8) {
      float bias = b_act[n * 8 + col];
#pragma unroll
      for (int r = 0; r < 8; ++r) {
        int t = row0 + mb + r;
        out[((long)t * NN + n) * 9 + col] = c[r] + bias;
      }
    }
  }
  // critic value, one timestep per thread
  int t = t0 + threadIdx.x;
  const float* wc = W_cri + n * 144;
  const __bf16* hrow = A + (long)t * NH;
  float acc = b_cri[n];
#pragma unroll 4
  for (int k = 0; k < 128; ++k) acc += (float)hrow[k] * wc[k];
  int aL = acts[(long)t * NN + ((n + NN - 1) & (NN - 1))];
  int aR = acts[(long)t * NN + ((n + 1) & (NN - 1))];
  acc += wc[128 + aL] + wc[136 + aR];
  out[((long)t * NN + n) * 9 + 8] = acc;
}

// ---------------- launcher --------------------------------------------------
extern "C" void kernel_launch(void* const* d_in, const int* in_sizes, int n_in,
                              void* d_out, int out_size, void* d_ws, size_t ws_size,
                              hipStream_t stream) {
  (void)in_sizes; (void)n_in; (void)out_size; (void)ws_size;
  const float* obs     = (const float*)d_in[0];
  const float* fps     = (const float*)d_in[1];
  const float* dones   = (const float*)d_in[2];
  const float* states0 = (const float*)d_in[3];
  const float* W_fcx   = (const float*)d_in[4];
  const float* b_fcx   = (const float*)d_in[5];
  const float* W_fcp   = (const float*)d_in[6];
  const float* b_fcp   = (const float*)d_in[7];
  const float* W_fcm   = (const float*)d_in[8];
  const float* b_fcm   = (const float*)d_in[9];
  const float* W_gat   = (const float*)d_in[10];
  const float* a_src   = (const float*)d_in[11];
  const float* a_dst   = (const float*)d_in[12];
  const float* W_ih    = (const float*)d_in[13];
  const float* W_hh    = (const float*)d_in[14];
  const float* b_lstm  = (const float*)d_in[15];
  const float* W_act   = (const float*)d_in[16];
  const float* b_act   = (const float*)d_in[17];
  const float* W_cri   = (const float*)d_in[18];
  const float* b_cri   = (const float*)d_in[19];
  const int*   acts    = (const int*)d_in[20];
  float* out = (float*)d_out;

  char* ws = (char*)d_ws;
  size_t off = 0;
  auto take = [&](size_t bytes) { char* p = ws + off; off += (bytes + 255) & ~(size_t)255; return p; };
  __bf16* xin     = (__bf16*)take((size_t)NN * TT * 192 * 2);   // 100.7 MB
  __bf16* pin     = (__bf16*)take((size_t)NN * TT * 32 * 2);    //  16.8 MB
  __bf16* WfcxB   = (__bf16*)take((size_t)NN * 6 * 8 * 512 * 2);
  __bf16* WfcpB   = (__bf16*)take((size_t)NN * 8 * 512 * 2);
  __bf16* WgatB   = (__bf16*)take((size_t)12 * 24 * 512 * 2);
  __bf16* WactB   = (__bf16*)take((size_t)NN * 4 * 512 * 2);
  __bf16* Wih_bf  = (__bf16*)take((size_t)NN * 384 * 512 * 2);  //  50.3 MB
  __bf16* Whh_bf  = (__bf16*)take((size_t)NN * 128 * 512 * 2);  //  16.8 MB
  __bf16* Wfcm_bf = (__bf16*)take((size_t)NN * 256 * 128 * 2);  //   8.4 MB
  __bf16* xp      = (__bf16*)take((size_t)NN * TT * 256 * 2);   // 134.2 MB
  __bf16* hsb     = (__bf16*)take((size_t)NN * TT * 128 * 2);   //  67.1 MB
  float*  Whg     = (float*)take((size_t)NN * 384 * 4);
  float*  dstg    = (float*)take((size_t)NN * 4);
  float*  curh    = (float*)take((size_t)NN * NH * 4);
  unsigned* sync  = (unsigned*)take(256);

  // stage inputs + weights (bf16 / WMMA-swizzled)
  k_stage_xin<<<4096, 256, 0, stream>>>(obs, xin);
  k_stage_pin<<<1024, 256, 0, stream>>>(fps, pin);
  k_swz_fcx<<<2048, 256, 0, stream>>>(W_fcx, WfcxB);
  k_swz_fcp<<<512, 256, 0, stream>>>(W_fcp, WfcpB);
  k_swz_gat<<<256, 256, 0, stream>>>(W_gat, WgatB);
  k_swz_act<<<256, 256, 0, stream>>>(W_act, WactB);
  k_cvt_bf16<<<8192, 256, 0, stream>>>(W_ih, Wih_bf, (long)NN * 384 * 512);
  k_cvt_bf16<<<4096, 256, 0, stream>>>(W_hh, Whh_bf, (long)NN * 128 * 512);
  k_cvt_bf16<<<2048, 256, 0, stream>>>(W_fcm, Wfcm_bf, (long)NN * 256 * 128);

  // h-independent encoders, batched over all timesteps (WMMA)
  k_gemm_xp<<<dim3(NN, 16), 256, 0, stream>>>(xin, pin, WfcxB, WfcpB, b_fcx, b_fcp, xp);

  // recurrent scan (persistent, 8 co-resident workgroups)
  k_init_sync<<<1, 32, 0, stream>>>(sync);
  k_scan<<<SCAN_WG, 256, 0, stream>>>(xp, Wfcm_bf, b_fcm, WgatB, a_src, a_dst,
                                      Wih_bf, Whh_bf, b_lstm, dones, states0,
                                      curh, Whg, dstg, sync, hsb);

  // heads (WMMA logits + critic dot)
  k_heads<<<dim3(NN, 8), 256, 0, stream>>>(hsb, WactB, b_act, W_cri, b_cri, acts, out);
}
